// TransformerBlock_22239340658874
// MI455X (gfx1250) — matrix operand order repaired and checked by reading, not compiled, not run on hardware
//
#include <hip/hip_runtime.h>

// ---------------------------------------------------------------------------
// Transformer block for MI455X (gfx1250): pre-LN attention + FFN.
// Matmuls: v_wmma_f32_16x16x32_bf16 (bf16 in, f32 accumulate).
// Data movement: Tensor Data Mover (tensor_load_to_lds, TENSORcnt) for GEMM
// A-tiles with LDS row padding + double buffering; global_load_async_to_lds
// (ASYNCcnt) for attention Q staging.
// ---------------------------------------------------------------------------

typedef __bf16 bf16_t;
typedef __attribute__((ext_vector_type(16))) __bf16 v16bf;
typedef __attribute__((ext_vector_type(8)))  __bf16 v8bf;
typedef __attribute__((ext_vector_type(8)))  float  v8f;
typedef unsigned int u32x4 __attribute__((ext_vector_type(4)));
typedef int          i32x4 __attribute__((ext_vector_type(4)));
typedef int          i32x8 __attribute__((ext_vector_type(8)));

#define BB   2
#define SS   2048
#define DD   1024
#define HH   16
#define DHH  64
#define DFFF 4096

#if __has_builtin(__builtin_amdgcn_tensor_load_to_lds)
#define HAVE_TDM 1
#else
#define HAVE_TDM 0
#endif

__device__ __forceinline__ bf16_t f2bf(float f) {
  unsigned u = __builtin_bit_cast(unsigned, f);
  u += 0x7FFFu + ((u >> 16) & 1u);            // round-to-nearest-even
  unsigned short h = (unsigned short)(u >> 16);
  return __builtin_bit_cast(bf16_t, h);
}

// Build a 16-element bf16 fragment from two 16-byte chunks.
__device__ __forceinline__ v16bf ldfrag(const bf16_t* p0, const bf16_t* p1) {
  union { v16bf v; v8bf h[2]; } u;
  u.h[0] = *(const v8bf*)p0;
  u.h[1] = *(const v8bf*)p1;
  return u.v;
}

__device__ __forceinline__ v8f wmma_bf16(v16bf a, v16bf b, v8f c) {
  return __builtin_amdgcn_wmma_f32_16x16x32_bf16(
      false, a, false, b, (short)0, c, false, false);
}

// --- CDNA5 async copy global -> LDS (per-lane, ASYNCcnt) -------------------
__device__ __forceinline__ void async_load_b128(unsigned lds_off,
                                                const void* gaddr) {
  asm volatile("global_load_async_to_lds_b128 %0, %1, off"
               :: "v"(lds_off), "v"(gaddr) : "memory");
}
__device__ __forceinline__ void wait_async0() {
#if __has_builtin(__builtin_amdgcn_s_wait_asynccnt)
  __builtin_amdgcn_s_wait_asynccnt(0);
#else
  asm volatile("s_wait_asynccnt 0x0" ::: "memory");
#endif
}
__device__ __forceinline__ void wait_tensor0() {
#if __has_builtin(__builtin_amdgcn_s_wait_tensorcnt)
  __builtin_amdgcn_s_wait_tensorcnt(0);
#else
  asm volatile("s_wait_tensorcnt 0x0" ::: "memory");
#endif
}

// --- Tensor Data Mover: DMA a 2D bf16 tile (tile_d0 x tile_d1) from global
// (row stride `stride0` elements) into LDS at lds_off, optionally padding
// LDS by 16B per 64B row (matches the 40-element LDS pitch). ---------------
__device__ __forceinline__ void tdm_load_2d_bf16(
    unsigned lds_off, const void* gptr,
    unsigned tensor_d0, unsigned tensor_d1,
    unsigned tile_d0, unsigned tile_d1,
    unsigned long long stride0, bool pad) {
#if HAVE_TDM
  unsigned long long ga = (unsigned long long)gptr;
  u32x4 g0;
  g0[0] = 1u;                                   // count=1, user descriptor
  g0[1] = lds_off;                              // LDS byte address
  g0[2] = (unsigned)(ga & 0xFFFFFFFFu);         // global_addr[31:0]
  g0[3] = (unsigned)((ga >> 32) & 0x01FFFFFFu)  // global_addr[56:32]
        | 0x80000000u;                          // type=2 ("image")
  union { unsigned u32[8]; unsigned long long u64[4]; } g1;
  unsigned long long w0 = (1ull << 16);         // data_size=1 -> 2 bytes
  if (pad) w0 |= (1ull << 20)                   // pad_enable
             |  (3ull << 22)                    // pad_interval: 16 DWORDs (64B)
             |  (3ull << 25);                   // pad_amount:   4 DWORDs (16B)
  w0 |= ((unsigned long long)(tensor_d0 & 0xFFFFu)) << 48;   // dim0[15:0]
  g1.u64[0] = w0;
  g1.u64[1] = (unsigned long long)(tensor_d0 >> 16)          // dim0[31:16]
            | ((unsigned long long)tensor_d1 << 16)          // dim1[31:0]
            | ((unsigned long long)(tile_d0 & 0xFFFFu) << 48);
  g1.u64[2] = (unsigned long long)(tile_d1 & 0xFFFFu)        // tile_dim1
            | ((stride0 & 0xFFFFFFFFull) << 32);             // stride0[31:0]
  g1.u64[3] = (stride0 >> 32) & 0xFFFFull;                   // stride0[47:32]
  i32x8 vg1;
  for (int i = 0; i < 8; ++i) vg1[i] = (int)g1.u32[i];
  i32x4 z4 = {0, 0, 0, 0};
#if __clang_major__ >= 23
  i32x8 z8 = {0, 0, 0, 0, 0, 0, 0, 0};
  __builtin_amdgcn_tensor_load_to_lds(g0, vg1, z4, z4, z8, 0);
#else
  __builtin_amdgcn_tensor_load_to_lds(g0, vg1, z4, z4, 0);
#endif
#endif
}

// ---------------------------------------------------------------------------
// fp32 -> bf16 conversion (weights)
// ---------------------------------------------------------------------------
__global__ __launch_bounds__(256) void f32_to_bf16_kernel(
    const float* __restrict__ in, bf16_t* __restrict__ out, int n) {
  int i = blockIdx.x * 256 + threadIdx.x;
  if (i < n) out[i] = f2bf(in[i]);
}

// ---------------------------------------------------------------------------
// LayerNorm over D=1024, one block per row, bf16 output.
// ---------------------------------------------------------------------------
__global__ __launch_bounds__(256) void ln_bf16_kernel(
    const float* __restrict__ x, const float* __restrict__ w,
    const float* __restrict__ b, bf16_t* __restrict__ out) {
  __shared__ float rs[256], rs2[256];
  int row = blockIdx.x;
  const float* xr = x + (size_t)row * DD;
  float s = 0.f, s2 = 0.f;
  for (int i = threadIdx.x; i < DD; i += 256) {
    float v = xr[i]; s += v; s2 += v * v;
  }
  rs[threadIdx.x] = s; rs2[threadIdx.x] = s2;
  __syncthreads();
  for (int o = 128; o > 0; o >>= 1) {
    if (threadIdx.x < o) {
      rs[threadIdx.x]  += rs[threadIdx.x + o];
      rs2[threadIdx.x] += rs2[threadIdx.x + o];
    }
    __syncthreads();
  }
  float mean = rs[0] * (1.f / DD);
  float var  = rs2[0] * (1.f / DD) - mean * mean;
  float inv  = rsqrtf(var + 1e-5f);
  for (int i = threadIdx.x; i < DD; i += 256)
    out[(size_t)row * DD + i] = f2bf((xr[i] - mean) * inv * w[i] + b[i]);
}

// ---------------------------------------------------------------------------
// V transpose: [B,S,H,DH] bf16 -> [B,H,DH,S] bf16.
// ---------------------------------------------------------------------------
__global__ __launch_bounds__(256) void transpose_v_kernel(
    const bf16_t* __restrict__ v, bf16_t* __restrict__ vT) {
  int idx = blockIdx.x * 256 + threadIdx.x;           // B*S*H*DH = 4M
  int d = idx & 63;
  int h = (idx >> 6) & 15;
  int s = (idx >> 10) & 2047;
  int b = idx >> 21;
  vT[(((size_t)(b * HH + h) * DHH + d) << 11) + s] = v[idx];
}

// ---------------------------------------------------------------------------
// bf16 WMMA GEMM: C[M,N] = A[M,K] * B[K,N] (+bias)(+res)(relu).
// Block tile 128x128, 8 waves x (64x32), BK=32, double-buffered LDS.
// A tile: one TDM tensor_load_to_lds per workgroup per K-step (row-padded).
// B tile: global->VGPR->LDS transposed so fragment K-runs are contiguous.
// ---------------------------------------------------------------------------
template <bool RELU, bool HAS_BIAS, bool HAS_RES, bool OUT_F32, bool OUT_BF16>
__global__ __launch_bounds__(256) void gemm_bf16_kernel(
    const bf16_t* __restrict__ A, size_t strideAz, int lda,
    const bf16_t* __restrict__ Bm, size_t strideBz, int ldb,
    const float* __restrict__ bias,
    const float* __restrict__ res,
    float* __restrict__ outF, bf16_t* __restrict__ outB,
    size_t strideCz, int ldc, int N, int K) {
  __shared__ bf16_t As[2][128 * 40];   // pitch 40 bf16 = 80 B (64B row + 16B pad)
  __shared__ bf16_t Bs[2][128 * 40];

  int z = blockIdx.z;
  A  += (size_t)z * strideAz;
  Bm += (size_t)z * strideBz;
  size_t coff = (size_t)z * strideCz;

  int bm = blockIdx.y * 128, bn = blockIdx.x * 128;
  int tid = threadIdx.x, lane = tid & 31, wave = tid >> 5;
  int wm = (wave & 1) * 64, wn = (wave >> 1) * 32;
  int mfr = lane & 15, khalf = lane >> 4;
  int kr = tid >> 3, grp = tid & 7, cb = bn + grp * 16;

  v8f acc[4][2] = {};
  bf16_t tmp[16];

  auto issueA = [&](int buf, int k0) {
#if HAVE_TDM
    if (wave == 0) {
      tdm_load_2d_bf16((unsigned)(size_t)&As[buf][0],
                       A + (size_t)bm * lda + k0,
                       (unsigned)K, 1u << 30, 32, 128,
                       (unsigned long long)lda, true);
    }
#else
    int r = tid >> 1, half = tid & 1;
    const bf16_t* ga = A + (size_t)(bm + r) * lda + k0 + half * 16;
    async_load_b128((unsigned)(size_t)&As[buf][r * 40 + half * 16], ga);
#endif
  };
  auto waitA = [&]() {
#if HAVE_TDM
    if (wave == 0) wait_tensor0();
#else
    wait_async0();
#endif
  };
  auto loadBregs = [&](int k0) {
    const bf16_t* gb = Bm + (size_t)(k0 + kr) * ldb + cb;
    if (cb + 15 < N) {
      *(v8bf*)tmp       = *(const v8bf*)gb;
      *(v8bf*)(tmp + 8) = *(const v8bf*)(gb + 8);
    } else {
      for (int e = 0; e < 16; ++e)
        tmp[e] = (cb + e < N) ? gb[e] : f2bf(0.f);
    }
  };
  auto storeBregs = [&](int buf) {
    for (int e = 0; e < 16; ++e) Bs[buf][(grp * 16 + e) * 40 + kr] = tmp[e];
  };

  // prologue: tile 0
  issueA(0, 0);
  loadBregs(0);
  storeBregs(0);
  waitA();
  __syncthreads();

  int nIter = K >> 5, cur = 0;
  for (int it = 0; it < nIter; ++it) {
    bool more = (it + 1) < nIter;
    if (more) {                       // prefetch next tile while computing
      issueA(cur ^ 1, (it + 1) << 5);
      loadBregs((it + 1) << 5);
    }
    v16bf af[4], bfr[2];
    for (int mt = 0; mt < 4; ++mt) {
      const bf16_t* pa = &As[cur][(wm + mt * 16 + mfr) * 40 + khalf * 8];
      af[mt] = ldfrag(pa, pa + 16);   // K: h*8..+7 and 16+h*8..+7
    }
    for (int nt = 0; nt < 2; ++nt) {
      const bf16_t* pb = &Bs[cur][(wn + nt * 16 + mfr) * 40 + khalf * 8];
      bfr[nt] = ldfrag(pb, pb + 16);  // K: h*8..+7 and 16+h*8..+7
    }
    for (int mt = 0; mt < 4; ++mt)
      for (int nt = 0; nt < 2; ++nt)
        acc[mt][nt] = wmma_bf16(af[mt], bfr[nt], acc[mt][nt]);
    if (more) storeBregs(cur ^ 1);
    waitA();
    __syncthreads();
    cur ^= 1;
  }

  // epilogue: C/D layout: VGPR r -> row r + 8*(lane/16), col lane%16
  for (int mt = 0; mt < 4; ++mt)
    for (int nt = 0; nt < 2; ++nt)
      for (int r = 0; r < 8; ++r) {
        int row = bm + wm + mt * 16 + r + ((lane >> 4) << 3);
        int col = bn + wn + nt * 16 + (lane & 15);
        if (col < N) {
          float v = acc[mt][nt][r];
          if constexpr (HAS_BIAS) v += bias[col];
          size_t oi = (size_t)row * ldc + coff + col;
          if constexpr (HAS_RES) v += res[oi];
          if constexpr (RELU) v = fmaxf(v, 0.f);
          if constexpr (OUT_F32)  outF[oi] = v;
          if constexpr (OUT_BF16) outB[oi] = f2bf(v);
        }
      }
}

// ---------------------------------------------------------------------------
// Causal attention: one workgroup (4 waves) per (b, h, 16-row block).
// Scores for the causal-truncated row block live entirely in LDS (<=128 KB,
// possible thanks to CDNA5's 320 KB WGP LDS), then softmax, then P@V WMMA
// against pre-transposed V.
// ---------------------------------------------------------------------------
__global__ __launch_bounds__(128) void attn_kernel(
    const bf16_t* __restrict__ q, const bf16_t* __restrict__ k,
    const bf16_t* __restrict__ vT, bf16_t* __restrict__ o) {
  extern __shared__ char smem[];
  float*  scores = (float*)smem;                                  // 16 x 2048 f32
  bf16_t* probs  = (bf16_t*)(smem + 16 * SS * 4);                 // 16 x 2048 bf16
  bf16_t* Qs     = (bf16_t*)(smem + 16 * SS * 4 + 16 * SS * 2);   // 16 x 72
  float*  red    = (float*)(Qs + 16 * 72);                        // 16 x 8

  int r0 = blockIdx.x * 16, h = blockIdx.y, b = blockIdx.z;
  int tid = threadIdx.x, lane = tid & 31, wave = tid >> 5;
  int nChunks = r0 / 16 + 1;                 // causal: only j <= r0+15 matters
  int jmax = nChunks * 16;

  // stage Q tile 16x64 via async copy: 128 threads x 16B
  {
    int i = tid * 8;
    int r = i >> 6, d = i & 63;
    async_load_b128((unsigned)(size_t)&Qs[r * 72 + d],
                    q + (((size_t)(b * SS + r0 + r) * HH + h) << 6) + d);
  }
  if (nChunks & 1) {                         // zero pad-chunk of probs
    for (int i = tid; i < 256; i += 128) {
      int r = i >> 4, c = i & 15;
      probs[r * SS + jmax + c] = f2bf(0.f);
    }
  }
  wait_async0();
  __syncthreads();

  // Q fragments (A matrix 16x64 -> two k32 steps)
  int m = lane & 15, khalf = lane >> 4;
  const bf16_t* qp = Qs + m * 72 + khalf * 8;
  v16bf aq0 = ldfrag(qp,      qp + 16);      // K(dh) 0..31
  v16bf aq1 = ldfrag(qp + 32, qp + 48);      // K(dh) 32..63

  // ---- scores: S_tile = Q (16x64) @ K^T chunk (64x16), scale + mask ----
  for (int jc = wave; jc < nChunks; jc += 4) {
    int j0 = jc * 16;
    const bf16_t* kp =
        k + (((size_t)(b * SS + j0 + (lane & 15)) * HH + h) << 6) + khalf * 8;
    v8f acc = {};
    acc = wmma_bf16(aq0, ldfrag(kp,      kp + 16), acc);
    acc = wmma_bf16(aq1, ldfrag(kp + 32, kp + 48), acc);
    for (int r = 0; r < 8; ++r) {
      int mrow = r + (khalf << 3);
      int jcol = j0 + (lane & 15);
      float sv = acc[r] * 0.125f;            // 1/sqrt(64)
      if (jcol > r0 + mrow) sv = -1e9f;      // causal mask
      scores[mrow * SS + jcol] = sv;
    }
  }
  __syncthreads();

  // ---- softmax: 8 threads per row ----
  {
    int row = tid >> 3, sub = tid & 7;
    float mx = -3.0e38f;
    for (int j = sub; j < jmax; j += 8) mx = fmaxf(mx, scores[row * SS + j]);
    red[row * 8 + sub] = mx;
    __syncthreads();
    mx = red[row * 8];
    for (int i = 1; i < 8; ++i) mx = fmaxf(mx, red[row * 8 + i]);
    __syncthreads();
    float sum = 0.f;
    for (int j = sub; j < jmax; j += 8) sum += __expf(scores[row * SS + j] - mx);
    red[row * 8 + sub] = sum;
    __syncthreads();
    sum = 0.f;
    for (int i = 0; i < 8; ++i) sum += red[row * 8 + i];
    float inv = 1.f / sum;
    for (int j = sub; j < jmax; j += 8)
      probs[row * SS + j] = f2bf(__expf(scores[row * SS + j] - mx) * inv);
  }
  __syncthreads();

  // ---- O = P (16 x jmax) @ V (jmax x 64) ----
  v8f oacc[4] = {};
  int nK32 = ((nChunks + 1) >> 1);
  for (int kc = wave; kc < nK32; kc += 4) {
    int k0 = kc * 32;
    const bf16_t* pp = probs + (lane & 15) * SS + k0 + khalf * 8;
    v16bf ap = ldfrag(pp, pp + 16);
    for (int nt = 0; nt < 4; ++nt) {
      int dh = nt * 16 + (lane & 15);
      const bf16_t* vp =
          vT + (((size_t)(b * HH + h) * DHH + dh) << 11) + k0 + (khalf << 3);
      oacc[nt] = wmma_bf16(ap, ldfrag(vp, vp + 16), oacc[nt]);
    }
  }
  __syncthreads();                            // done with scores region
  float* part = scores;                       // [4][16][64] partials
  for (int nt = 0; nt < 4; ++nt)
    for (int r = 0; r < 8; ++r)
      part[(wave * 16 + r + (khalf << 3)) * 64 + nt * 16 + (lane & 15)] =
          oacc[nt][r];
  __syncthreads();
  for (int i = tid; i < 16 * 64; i += 128) {
    int r = i >> 6, d = i & 63;
    float s = part[r * 64 + d] + part[(16 + r) * 64 + d] +
              part[(32 + r) * 64 + d] + part[(48 + r) * 64 + d];
    o[(((size_t)(b * SS + r0 + r) * HH + h) << 6) + d] = f2bf(s);
  }
}

// ---------------------------------------------------------------------------
// Host side
// ---------------------------------------------------------------------------
extern "C" void kernel_launch(void* const* d_in, const int* in_sizes, int n_in,
                              void* d_out, int out_size, void* d_ws,
                              size_t ws_size, hipStream_t stream) {
  const float* x    = (const float*)d_in[0];
  // d_in[1] = causal mask (tril) — causality hardcoded in attn_kernel.
  const float* Wq   = (const float*)d_in[2];
  const float* Wk   = (const float*)d_in[3];
  const float* Wv   = (const float*)d_in[4];
  const float* Wo   = (const float*)d_in[5];
  const float* bo   = (const float*)d_in[6];
  const float* W1   = (const float*)d_in[7];
  const float* b1   = (const float*)d_in[8];
  const float* W2   = (const float*)d_in[9];
  const float* b2   = (const float*)d_in[10];
  const float* ln1w = (const float*)d_in[11];
  const float* ln1b = (const float*)d_in[12];
  const float* ln2w = (const float*)d_in[13];
  const float* ln2b = (const float*)d_in[14];

  const int M = BB * SS;                      // 4096 token rows

  char* ws = (char*)d_ws;
  size_t off = 0;
  auto alloc = [&](size_t bytes) {
    size_t o = off;
    off += (bytes + 255) & ~(size_t)255;
    return o;
  };
  bf16_t* xn  = (bf16_t*)(ws + alloc((size_t)M * DD * 2));
  bf16_t* Wqb = (bf16_t*)(ws + alloc((size_t)DD * DD * 2));
  bf16_t* Wkb = (bf16_t*)(ws + alloc((size_t)DD * DD * 2));
  bf16_t* Wvb = (bf16_t*)(ws + alloc((size_t)DD * DD * 2));
  bf16_t* Wob = (bf16_t*)(ws + alloc((size_t)DD * DD * 2));
  bf16_t* W1b = (bf16_t*)(ws + alloc((size_t)DD * DFFF * 2));
  bf16_t* W2b = (bf16_t*)(ws + alloc((size_t)DFFF * DD * 2));
  bf16_t* qb  = (bf16_t*)(ws + alloc((size_t)M * DD * 2));   // [B,S,H,DH]
  bf16_t* kb  = (bf16_t*)(ws + alloc((size_t)M * DD * 2));
  bf16_t* vb  = (bf16_t*)(ws + alloc((size_t)M * DD * 2));
  bf16_t* vTb = (bf16_t*)(ws + alloc((size_t)M * DD * 2));   // [B,H,DH,S]
  bf16_t* ob  = (bf16_t*)(ws + alloc((size_t)M * DD * 2));   // attn out
  float*  x1  = (float*) (ws + alloc((size_t)M * DD * 4));   // x + attn_out
  bf16_t* xn2 = (bf16_t*)(ws + alloc((size_t)M * DD * 2));
  bf16_t* hb  = (bf16_t*)(ws + alloc((size_t)M * DFFF * 2)); // relu(ffn1)

  f32_to_bf16_kernel<<<(DD * DD + 255) / 256, 256, 0, stream>>>(Wq, Wqb, DD * DD);
  f32_to_bf16_kernel<<<(DD * DD + 255) / 256, 256, 0, stream>>>(Wk, Wkb, DD * DD);
  f32_to_bf16_kernel<<<(DD * DD + 255) / 256, 256, 0, stream>>>(Wv, Wvb, DD * DD);
  f32_to_bf16_kernel<<<(DD * DD + 255) / 256, 256, 0, stream>>>(Wo, Wob, DD * DD);
  f32_to_bf16_kernel<<<(DD * DFFF + 255) / 256, 256, 0, stream>>>(W1, W1b, DD * DFFF);
  f32_to_bf16_kernel<<<(DFFF * DD + 255) / 256, 256, 0, stream>>>(W2, W2b, DFFF * DD);

  ln_bf16_kernel<<<M, 256, 0, stream>>>(x, ln1w, ln1b, xn);

  // Q/K/V projections: per-head GEMMs (Z=16, N=64, K=1024)
  {
    dim3 g(1, M / 128, HH);
    gemm_bf16_kernel<false, false, false, false, true><<<g, 256, 0, stream>>>(
        xn, 0, DD, Wqb, (size_t)DD * DHH, DHH, nullptr, nullptr, nullptr, qb,
        DHH, DD, DHH, DD);
    gemm_bf16_kernel<false, false, false, false, true><<<g, 256, 0, stream>>>(
        xn, 0, DD, Wkb, (size_t)DD * DHH, DHH, nullptr, nullptr, nullptr, kb,
        DHH, DD, DHH, DD);
    gemm_bf16_kernel<false, false, false, false, true><<<g, 256, 0, stream>>>(
        xn, 0, DD, Wvb, (size_t)DD * DHH, DHH, nullptr, nullptr, nullptr, vb,
        DHH, DD, DHH, DD);
  }

  transpose_v_kernel<<<(M * DD + 255) / 256, 256, 0, stream>>>(vb, vTb);

  // attention (causal, LDS-resident scores)
  {
    dim3 g(SS / 16, HH, BB);
    size_t shmem = 16 * SS * 4 + 16 * SS * 2 + 16 * 72 * 2 + 16 * 8 * 4;
    attn_kernel<<<g, 128, shmem, stream>>>(qb, kb, vTb, ob);
  }

  // output projection + bias + residual -> x1 (f32)
  {
    dim3 g(DD / 128, M / 128, 1);
    gemm_bf16_kernel<false, true, true, true, false><<<g, 256, 0, stream>>>(
        ob, 0, DD, Wob, 0, DD, bo, x, x1, nullptr, 0, DD, DD, DD);
  }

  ln_bf16_kernel<<<M, 256, 0, stream>>>(x1, ln2w, ln2b, xn2);

  // FFN1 + bias + relu -> hb (bf16)
  {
    dim3 g(DFFF / 128, M / 128, 1);
    gemm_bf16_kernel<true, true, false, false, true><<<g, 256, 0, stream>>>(
        xn2, 0, DD, W1b, 0, DFFF, b1, nullptr, nullptr, hb, 0, DFFF, DFFF, DD);
  }

  // FFN2 + bias + residual -> d_out (f32)
  {
    dim3 g(DD / 128, M / 128, 1);
    gemm_bf16_kernel<false, true, true, true, false><<<g, 256, 0, stream>>>(
        hb, 0, DFFF, W2b, 0, DD, b2, x1, (float*)d_out, nullptr, 0, DD, DD,
        DFFF);
  }
}
